// XnorBlock_83726092468778
// MI455X (gfx1250) — 8-bit matrix operand order repaired and checked by reading, compiled, not run on hardware
//
#include <hip/hip_runtime.h>
#include <cstdint>

#define NB 32
#define CH 256
#define HS 56
#define HW 3136
#define NP 100352
#define EPSF 1e-5f

#ifndef __has_builtin
#define __has_builtin(x) 0
#endif
#if __has_builtin(__builtin_amdgcn_global_load_async_to_lds_b128) && \
    __has_builtin(__builtin_amdgcn_s_wait_asynccnt)
#define USE_ASYNC_WB 1
#else
#define USE_ASYNC_WB 0
#endif

typedef __attribute__((ext_vector_type(8))) int v8i;

#if USE_ASYNC_WB
typedef int vi4 __attribute__((ext_vector_type(4)));
typedef __attribute__((address_space(1))) vi4 g_vi4;
typedef __attribute__((address_space(3))) vi4 l_vi4;

__device__ __forceinline__ void async_cp16(const uint8_t* src, uint8_t* ldsDst) {
    __builtin_amdgcn_global_load_async_to_lds_b128(
        (g_vi4*)(void*)const_cast<uint8_t*>(src),
        (l_vi4*)(void*)ldsDst, 0, 0);
}
#endif

__device__ __forceinline__ int8_t sgn8(float v) {
    return v > 0.f ? (int8_t)1 : (v < 0.f ? (int8_t)-1 : (int8_t)0);
}

__global__ __launch_bounds__(256) void k_bn1_stats(
    const float* __restrict__ x, const float* __restrict__ gamma,
    const float* __restrict__ beta, float* __restrict__ scale,
    float* __restrict__ shift)
{
    __shared__ float rs[256], rq[256];
    int c = blockIdx.x, t = threadIdx.x;
    float s = 0.f, q = 0.f;
    for (int i = t; i < NP; i += 256) {
        int n = i / HW;
        int hw = i - n * HW;
        float v = x[((size_t)n * CH + c) * HW + hw];
        s += v; q += v * v;
    }
    rs[t] = s; rq[t] = q;
    __syncthreads();
    for (int st = 128; st > 0; st >>= 1) {
        if (t < st) { rs[t] += rs[t + st]; rq[t] += rq[t + st]; }
        __syncthreads();
    }
    if (t == 0) {
        float mean = rs[0] * (1.f / (float)NP);
        float var  = rq[0] * (1.f / (float)NP) - mean * mean;
        float sc   = gamma[c] * rsqrtf(var + EPSF);
        scale[c] = sc;
        shift[c] = beta[c] - mean * sc;
    }
}

__global__ __launch_bounds__(256) void k_wprep(
    const float* __restrict__ w1, const float* __restrict__ w2,
    int8_t* __restrict__ f1, int8_t* __restrict__ f2,
    float* __restrict__ a1, float* __restrict__ a2)
{
    const float* w = blockIdx.y ? w2 : w1;
    int8_t* f = blockIdx.y ? f2 : f1;
    float* al = blockIdx.y ? a2 : a1;
    int co = blockIdx.x, ci = threadIdx.x;
    __shared__ float red[256];
    float s = 0.f;
    for (int tap = 0; tap < 9; ++tap) {
        float v = w[((size_t)co * CH + ci) * 9 + tap];
        s += fabsf(v);
        int kc   = ci >> 6;
        int r    = ci & 63;
        int half = r >> 4;
        int lane = (co & 15) + ((r & 8) ? 16 : 0);
        int j    = r & 7;
        size_t frag = ((size_t)(co >> 4) * 9 + tap) * 4 + kc;
        f[(frag * 32 + lane) * 32 + half * 8 + j] = sgn8(v);
    }
    red[ci] = s;
    __syncthreads();
    for (int st = 128; st > 0; st >>= 1) {
        if (ci < st) red[ci] += red[ci + st];
        __syncthreads();
    }
    if (ci == 0) al[co] = red[0] * (1.f / 2304.f);
}

__global__ void k_zeropad(uint4* __restrict__ p1, uint4* __restrict__ p2)
{
    int t = threadIdx.x;
    uint4 z = make_uint4(0u, 0u, 0u, 0u);
    if (t < 16) p1[t] = z;
    else        p2[t - 16] = z;
}

__global__ __launch_bounds__(256) void k_sign1(
    const float* __restrict__ x, const float* __restrict__ scale,
    const float* __restrict__ shift, uint32_t* __restrict__ signOut,
    float* __restrict__ aOut)
{
    __shared__ int8_t sm[64 * 256];
    __shared__ float red[256];
    int t = threadIdx.x;
    int pw = t & 63, q = t >> 6;
    int pbase = blockIdx.x * 64;
    int n = pbase / HW;
    size_t rowaddr = (size_t)n * CH * HW + (size_t)(pbase + pw - n * HW);
    float acc = 0.f;
#pragma unroll 4
    for (int it = 0; it < 64; ++it) {
        int c = it * 4 + q;
        float v = x[rowaddr + (size_t)c * HW] * scale[c] + shift[c];
        acc += v;
        int c4s = (it + pw) & 63;
        sm[pw * 256 + c4s * 4 + q] = sgn8(v);
    }
    red[t] = acc;
    __syncthreads();
    if (t < 64)
        aOut[pbase + t] = (red[t] + red[t + 64] + red[t + 128] + red[t + 192]) * (1.f / 256.f);
#pragma unroll
    for (int it = 0; it < 16; ++it) {
        int idx = it * 256 + t;
        int ppw = idx >> 6, c4 = idx & 63;
        int c4s = (c4 + ppw) & 63;
        uint32_t word = *(const uint32_t*)&sm[ppw * 256 + c4s * 4];
        signOut[(size_t)(pbase + ppw) * 64 + c4] = word;
    }
}

__global__ __launch_bounds__(256) void k_boxK(
    const float* __restrict__ a, float* __restrict__ K)
{
    int p = blockIdx.x * 256 + threadIdx.x;
    int n = p / HW;
    int hw = p - n * HW;
    int h = hw / HS, w = hw - h * HS;
    float s = 0.f;
#pragma unroll
    for (int dy = -1; dy <= 1; ++dy)
#pragma unroll
        for (int dx = -1; dx <= 1; ++dx) {
            int hh = h + dy, ww = w + dx;
            if ((unsigned)hh < (unsigned)HS && (unsigned)ww < (unsigned)HS)
                s += a[(size_t)n * HW + hh * HS + ww];
        }
    K[p] = s * (1.f / 9.f);
}

template <int MODE>
__global__ __launch_bounds__(256) void k_binconv(
    const uint8_t* __restrict__ signIn,
    const uint8_t* __restrict__ wfrag,
    const float* __restrict__ alpha,
    const float* __restrict__ Kbuf,
    float* __restrict__ outBuf,
    const float* __restrict__ identity)
{
#if USE_ASYNC_WB
    __shared__ uint8_t wbuf[2][16384];
#endif
    const int tid  = threadIdx.x;
    const int lane = tid & 31;
    const int wv   = tid >> 5;
    const int ps   = wv >> 1;
    const int cs   = wv & 1;
    const int p0   = blockIdx.x * 128;
    const int co0  = blockIdx.y * 64;
    const int wavePixBase = p0 + ps * 32;
    const int waveCoBase  = co0 + cs * 32;
    const int lm = lane & 15;
    const bool hiHalf = lane >= 16;
    const int kOffLane = hiHalf ? 8 : 0;
    const int fragCo0 = waveCoBase >> 4;
    const int c16b = blockIdx.y * 4;

    int pn[2], ph[2], pww[2];
#pragma unroll
    for (int j = 0; j < 2; ++j) {
        int p = wavePixBase + j * 16 + lm;
        pn[j] = p / HW;
        int hw = p - pn[j] * HW;
        ph[j] = hw / HS;
        pww[j] = hw - ph[j] * HS;
    }

    v8i acc[2][2] = {};

#if USE_ASYNC_WB
#pragma unroll
    for (int i = 0; i < 4; ++i) {
        const uint8_t* src = wfrag + ((size_t)(c16b + i) * 9) * 4096 + (size_t)tid * 16;
        async_cp16(src, &wbuf[0][i * 4096 + tid * 16]);
    }
    __builtin_amdgcn_s_wait_asynccnt(0);
    __syncthreads();
#endif

    for (int tap = 0; tap < 9; ++tap) {
        const int dy = tap / 3 - 1, dx = tap % 3 - 1;

#if USE_ASYNC_WB
        if (tap < 8) {
#pragma unroll
            for (int i = 0; i < 4; ++i) {
                const uint8_t* src =
                    wfrag + ((size_t)(c16b + i) * 9 + (tap + 1)) * 4096 + (size_t)tid * 16;
                async_cp16(src, &wbuf[(tap + 1) & 1][i * 4096 + tid * 16]);
            }
        }
#else
        if (tap < 8)
            __builtin_prefetch(wfrag + ((size_t)fragCo0 * 9 + tap + 1) * 4096 +
                               (size_t)tid * 16, 0, 3);
#endif

        const uint8_t* rowA[2];
#pragma unroll
        for (int j = 0; j < 2; ++j) {
            int hh = ph[j] + dy, ww = pww[j] + dx;
            bool valid = ((unsigned)hh < (unsigned)HS) & ((unsigned)ww < (unsigned)HS);
            int prow = valid ? (pn[j] * HW + hh * HS + ww) : NP;
            rowA[j] = signIn + (size_t)prow * CH;
        }

#pragma unroll
        for (int kc = 0; kc < 4; ++kc) {
            const int kbase = kc * 64 + kOffLane;
            union { v8i v; uint2 q[4]; } A[2];
#pragma unroll
            for (int j = 0; j < 2; ++j) {
                const uint8_t* pA = rowA[j] + kbase;
                A[j].q[0] = *(const uint2*)(pA);
                A[j].q[1] = *(const uint2*)(pA + 16);
                A[j].q[2] = *(const uint2*)(pA + 32);
                A[j].q[3] = *(const uint2*)(pA + 48);
            }
            union { v8i v; uint4 q[2]; } B[2];
#pragma unroll
            for (int m = 0; m < 2; ++m) {
#if USE_ASYNC_WB
                const uint8_t* pB =
                    &wbuf[tap & 1][(cs * 2 + m) * 4096 + kc * 1024 + lane * 32];
#else
                size_t frag = ((size_t)(fragCo0 + m) * 9 + tap) * 4 + kc;
                const uint8_t* pB = wfrag + frag * 1024 + (size_t)lane * 32;
#endif
                B[m].q[0] = *(const uint4*)(pB);
                B[m].q[1] = *(const uint4*)(pB + 16);
            }
#pragma unroll
            for (int j = 0; j < 2; ++j)
#pragma unroll
                for (int m = 0; m < 2; ++m)
                    acc[j][m] = __builtin_amdgcn_wmma_i32_16x16x64_iu8(
                        true, A[j].v, true, B[m].v, acc[j][m], false, false);
        }

#if USE_ASYNC_WB
        __builtin_amdgcn_s_wait_asynccnt(0);
        __syncthreads();
#endif
    }

    if (MODE == 1) {
#pragma unroll
        for (int j = 0; j < 2; ++j) {
            const int pb = wavePixBase + j * 16 + (hiHalf ? 8 : 0);
            float Kv[8];
#pragma unroll
            for (int r = 0; r < 8; ++r) Kv[r] = Kbuf[pb + r];
#pragma unroll
            for (int m = 0; m < 2; ++m) {
                const int co = waveCoBase + m * 16 + lm;
                const float al = alpha[co];
#pragma unroll
                for (int r = 0; r < 8; ++r) {
                    float v = (float)acc[j][m][r] * al * Kv[r];
                    v = v > 0.f ? v : 0.f;
                    outBuf[(size_t)(pb + r) * CH + co] = v;
                }
            }
        }
    } else {
        __shared__ float smT[8 * 32 * 33];
        float* sm = smT + wv * (32 * 33);
#pragma unroll
        for (int j = 0; j < 2; ++j) {
            const int pl = j * 16 + (hiHalf ? 8 : 0);
            const int pbg = wavePixBase + pl;
            float Kv[8];
#pragma unroll
            for (int r = 0; r < 8; ++r) Kv[r] = Kbuf[pbg + r];
#pragma unroll
            for (int m = 0; m < 2; ++m) {
                const int co = waveCoBase + m * 16 + lm;
                const float al = alpha[co];
                const int cl = m * 16 + lm;
#pragma unroll
                for (int r = 0; r < 8; ++r)
                    sm[(pl + r) * 33 + cl] = (float)acc[j][m][r] * al * Kv[r];
            }
        }
        __syncthreads();
        const int p = wavePixBase + lane;
        const int n = p / HW;
        const int hw = p - n * HW;
#pragma unroll 4
        for (int cl = 0; cl < 32; ++cl) {
            const int co = waveCoBase + cl;
            const size_t addr = ((size_t)n * CH + co) * HW + hw;
            float v = sm[lane * 33 + cl] + identity[addr];
            v = v > 0.f ? v : 0.f;
            outBuf[addr] = v;
        }
    }
}

__global__ __launch_bounds__(256) void k_bn2_partial(
    const float* __restrict__ mid, float* __restrict__ ps, float* __restrict__ pq)
{
    int t = threadIdx.x, b = blockIdx.x;
    float s = 0.f, q = 0.f;
    size_t base = (size_t)b * 256 * CH + t;
    for (int i = 0; i < 256; ++i) {
        float v = mid[base + (size_t)i * CH];
        s += v; q += v * v;
    }
    ps[b * 256 + t] = s;
    pq[b * 256 + t] = q;
}

__global__ __launch_bounds__(256) void k_bn2_final(
    const float* __restrict__ ps, const float* __restrict__ pq,
    const float* __restrict__ gamma, const float* __restrict__ beta,
    float* __restrict__ scale, float* __restrict__ shift)
{
    int c = threadIdx.x;
    float s = 0.f, q = 0.f;
    for (int b = 0; b < NP / 256; ++b) {
        s += ps[b * 256 + c];
        q += pq[b * 256 + c];
    }
    float mean = s * (1.f / (float)NP);
    float var  = q * (1.f / (float)NP) - mean * mean;
    float sc   = gamma[c] * rsqrtf(var + EPSF);
    scale[c] = sc;
    shift[c] = beta[c] - mean * sc;
}

__global__ __launch_bounds__(256) void k_sign2(
    const float* __restrict__ mid, const float* __restrict__ scale,
    const float* __restrict__ shift, uint32_t* __restrict__ signOut,
    float* __restrict__ aOut)
{
    __shared__ float red[256];
    int t = threadIdx.x;
    int c4 = t & 63, pl = t >> 6;
    int p = blockIdx.x * 4 + pl;
    const float4 mv = *(const float4*)&mid[(size_t)p * CH + c4 * 4];
    int c = c4 * 4;
    float v0 = mv.x * scale[c + 0] + shift[c + 0];
    float v1 = mv.y * scale[c + 1] + shift[c + 1];
    float v2 = mv.z * scale[c + 2] + shift[c + 2];
    float v3 = mv.w * scale[c + 3] + shift[c + 3];
    uint32_t word = (uint32_t)(uint8_t)sgn8(v0)
                  | ((uint32_t)(uint8_t)sgn8(v1) << 8)
                  | ((uint32_t)(uint8_t)sgn8(v2) << 16)
                  | ((uint32_t)(uint8_t)sgn8(v3) << 24);
    signOut[(size_t)p * 64 + c4] = word;
    red[t] = v0 + v1 + v2 + v3;
    __syncthreads();
    for (int st = 32; st > 0; st >>= 1) {
        if (c4 < st) red[t] += red[t + st];
        __syncthreads();
    }
    if (c4 == 0) aOut[p] = red[pl * 64] * (1.f / 256.f);
}

extern "C" void kernel_launch(void* const* d_in, const int* in_sizes, int n_in,
                              void* d_out, int out_size, void* d_ws, size_t ws_size,
                              hipStream_t stream)
{
    const float* x  = (const float*)d_in[0];
    const float* g1 = (const float*)d_in[1];
    const float* b1 = (const float*)d_in[2];
    const float* w1 = (const float*)d_in[3];
    const float* g2 = (const float*)d_in[4];
    const float* b2 = (const float*)d_in[5];
    const float* w2 = (const float*)d_in[6];
    float* out = (float*)d_out;

    uint8_t* ws = (uint8_t*)d_ws;
    size_t off = 0;
    auto alloc = [&](size_t bytes) {
        uint8_t* p = ws + off;
        off += (bytes + 255) & ~(size_t)255;
        return p;
    };
    float*    scale1 = (float*)alloc(CH * 4);
    float*    shift1 = (float*)alloc(CH * 4);
    float*    scale2 = (float*)alloc(CH * 4);
    float*    shift2 = (float*)alloc(CH * 4);
    float*    alpha1 = (float*)alloc(CH * 4);
    float*    alpha2 = (float*)alloc(CH * 4);
    float*    k1buf  = (float*)alloc((size_t)NP * 4);
    float*    k2buf  = (float*)alloc((size_t)NP * 4);
    float*    abuf   = (float*)alloc((size_t)NP * 4);
    int8_t*   wfrag1 = (int8_t*)alloc((size_t)CH * 2304);
    int8_t*   wfrag2 = (int8_t*)alloc((size_t)CH * 2304);
    uint8_t*  sign1  = (uint8_t*)alloc((size_t)(NP + 1) * CH);
    uint8_t*  sign2  = (uint8_t*)alloc((size_t)(NP + 1) * CH);
    float*    mid    = (float*)alloc((size_t)NP * CH * 4);
    float*    psum   = (float*)alloc((size_t)(NP / 256) * CH * 4);
    float*    psq    = (float*)alloc((size_t)(NP / 256) * CH * 4);
    if (off > ws_size) return;

    k_bn1_stats<<<CH, 256, 0, stream>>>(x, g1, b1, scale1, shift1);
    k_wprep<<<dim3(CH, 2), 256, 0, stream>>>(w1, w2, wfrag1, wfrag2, alpha1, alpha2);
    k_zeropad<<<1, 32, 0, stream>>>((uint4*)(sign1 + (size_t)NP * CH),
                                    (uint4*)(sign2 + (size_t)NP * CH));
    k_sign1<<<NP / 64, 256, 0, stream>>>(x, scale1, shift1, (uint32_t*)sign1, abuf);
    k_boxK<<<NP / 256, 256, 0, stream>>>(abuf, k1buf);
    k_binconv<1><<<dim3(NP / 128, 4), 256, 0, stream>>>(
        sign1, (const uint8_t*)wfrag1, alpha1, k1buf, mid, nullptr);
    k_bn2_partial<<<NP / 256, 256, 0, stream>>>(mid, psum, psq);
    k_bn2_final<<<1, 256, 0, stream>>>(psum, psq, g2, b2, scale2, shift2);
    k_sign2<<<NP / 4, 256, 0, stream>>>(mid, scale2, shift2, (uint32_t*)sign2, abuf);
    k_boxK<<<NP / 256, 256, 0, stream>>>(abuf, k2buf);
    k_binconv<2><<<dim3(NP / 128, 4), 256, 0, stream>>>(
        sign2, (const uint8_t*)wfrag2, alpha2, k2buf, out, x);
}
